// MultiHeadAttentionRoPE_1975684956815
// MI455X (gfx1250) — matrix operand order repaired and checked by reading, not compiled, not run on hardware
//
#include <hip/hip_runtime.h>
#include <hip/hip_bf16.h>

// ---------------- problem constants ----------------
#define BB 4
#define SQ 2048
#define NH 16
#define HD 64
#define MD 1024   // NH*HD

typedef __bf16 bf16_t;
typedef __attribute__((ext_vector_type(16))) __bf16 v16bf;
typedef __attribute__((ext_vector_type(2)))  __bf16 v2bf;
typedef __attribute__((ext_vector_type(8)))  float  v8f;
typedef __attribute__((ext_vector_type(4)))  unsigned int v4u;
typedef int v4i_ __attribute__((vector_size(16)));   // matches async builtin

union Frag16 { v16bf v; v4u u[2]; };

// ---------------- bf16 conversion helpers ----------------
__device__ __forceinline__ bf16_t f2bf(float f) {
    unsigned u = __builtin_bit_cast(unsigned, f);
    unsigned r = u + 0x7FFFu + ((u >> 16) & 1u);   // round-to-nearest-even
    unsigned short h = (unsigned short)(r >> 16);
    return __builtin_bit_cast(bf16_t, h);
}

// pack two floats -> {lo,hi} bf16 in one dword (HW v_cvt_pk_bf16_f32 if present)
__device__ __forceinline__ unsigned pk2bf(float lo, float hi) {
#if __has_builtin(__builtin_amdgcn_cvt_pk_bf16_f32)
    v2bf v = __builtin_amdgcn_cvt_pk_bf16_f32(lo, hi);
    return __builtin_bit_cast(unsigned, v);
#else
    unsigned a = __builtin_bit_cast(unsigned, lo);
    unsigned b = __builtin_bit_cast(unsigned, hi);
    a = (a + 0x7FFFu + ((a >> 16) & 1u)) >> 16;
    b = (b + 0x7FFFu + ((b >> 16) & 1u)) & 0xFFFF0000u;
    return a | b;
#endif
}

// ---------------- CDNA5 async global->LDS (ASYNCcnt) ----------------
#if __has_builtin(__builtin_amdgcn_global_load_async_to_lds_b128)
#define HAVE_ASYNC_LDS 1
#else
#define HAVE_ASYNC_LDS 0
#endif

__device__ __forceinline__ void async_copy16(void* lds, const void* gsrc) {
#if HAVE_ASYNC_LDS
    __builtin_amdgcn_global_load_async_to_lds_b128(
        (__attribute__((address_space(1))) v4i_*)gsrc,
        (__attribute__((address_space(3))) v4i_*)lds, 0, 0);
#else
    *(v4u*)lds = *(const v4u*)gsrc;
#endif
}
__device__ __forceinline__ void async_wait() {
#if HAVE_ASYNC_LDS
#if __has_builtin(__builtin_amdgcn_s_wait_asynccnt)
    __builtin_amdgcn_s_wait_asynccnt(0);
#else
    asm volatile("s_wait_asynccnt 0" ::: "memory");
#endif
#endif
}

// =====================================================================
// Kernel 1: QKV projection GEMM (fp32 -> bf16 staged), fused bias + RoPE.
//   A = x [8192, 1024] fp32 (row = b*S+s), B = W_qkv [1024, 3072] fp32.
//   Epilogue writes q/k/v as bf16 in [B,H,S,D]; RoPE + 1/sqrt(D) folded
//   into q, RoPE into k.  Block tile 64x128, 8 waves (wave = 16x64).
// =====================================================================
__global__ __launch_bounds__(256) void qkv_rope_kernel(
    const float* __restrict__ X, const float* __restrict__ Wqkv,
    const float* __restrict__ bqkv, const float* __restrict__ cosT,
    const float* __restrict__ sinT,
    bf16_t* __restrict__ qB, bf16_t* __restrict__ kB, bf16_t* __restrict__ vB)
{
    constexpr int K = MD;
    constexpr int N = 3 * MD;
    constexpr int BK = 32, BM = 64, BN = 128;
    constexpr int LDA = BK + 8;   // elems
    constexpr int LDB = BK + 8;   // Bt stored transposed: [n][k]
    __shared__ bf16_t At[BM * LDA];
    __shared__ bf16_t Bt[BN * LDB];

    const int tid  = threadIdx.x;
    const int wave = tid >> 5;
    const int lane = tid & 31;
    const int half = lane >> 4;
    const int l15  = lane & 15;
    const int wr   = wave >> 1;          // 0..3 (M dir)
    const int wc   = wave & 1;           // 0..1 (N dir)
    const int rowBase = blockIdx.y * BM;
    const int colBase = blockIdx.x * BN;

    v8f acc[4] = {};

    for (int k0 = 0; k0 < K; k0 += BK) {
        __syncthreads();
        { // stage A tile (64x32 fp32 -> bf16): pack 8 floats -> one b128
            int r = tid >> 2;
            int c = (tid & 3) * 8;
            const float* src = X + (size_t)(rowBase + r) * K + k0 + c;
            v4u pack;
            pack.x = pk2bf(src[0], src[1]);
            pack.y = pk2bf(src[2], src[3]);
            pack.z = pk2bf(src[4], src[5]);
            pack.w = pk2bf(src[6], src[7]);
            *(v4u*)&At[r * LDA + c] = pack;
        }
        { // stage B tile (32x128 fp32) transposed -> Bt[n][k]; two k-rows
          // per thread so each transpose store is a packed b32
            int kp = tid >> 4;            // k = 2kp, 2kp+1
            int n0 = (tid & 15) * 8;
            const float* s0 = Wqkv + (size_t)(k0 + 2 * kp) * N + colBase + n0;
            const float* s1 = s0 + N;
            #pragma unroll
            for (int i = 0; i < 8; ++i)
                *(unsigned*)&Bt[(n0 + i) * LDB + 2 * kp] = pk2bf(s0[i], s1[i]);
        }
        __syncthreads();

        Frag16 a;   // A 16x32: lane-half splits K {0-7,16-23} vs {8-15,24-31}
        {
            int r = wr * 16 + l15;
            a.u[0] = *(const v4u*)&At[r * LDA + half * 8];
            a.u[1] = *(const v4u*)&At[r * LDA + 16 + half * 8];
        }
        #pragma unroll
        for (int t = 0; t < 4; ++t) {
            Frag16 b;  // B 32x16: lane n = column, K half by lane-half
            int n = wc * 64 + t * 16 + l15;
            b.u[0] = *(const v4u*)&Bt[n * LDB + half * 8];
            b.u[1] = *(const v4u*)&Bt[n * LDB + 16 + half * 8];
            acc[t] = __builtin_amdgcn_wmma_f32_16x16x32_bf16(
                false, a.v, false, b.v, (short)0, acc[t], false, false);
        }
    }

    // ---- epilogue: bias + RoPE, scatter to q/k/v [B,H,S,D] bf16 ----
    #pragma unroll
    for (int t = 0; t < 4; ++t) {
        const int colB = colBase + wc * 64 + t * 16;     // 16-col tile stays
        const int col  = colB + l15;                     //  inside one 64-blk
        const int h     = colB / 192;                    // head (uniform/tile)
        const int rem   = colB % 192;
        const int which = rem / 64;                      // 0=q 1=k 2=v
        const int d     = (rem % 64) + l15;
        const float bias = bqkv[col];
        float vals[8];
        #pragma unroll
        for (int r = 0; r < 8; ++r) vals[r] = acc[t][r] + bias;

        if (which == 2) {
            #pragma unroll
            for (int r = 0; r < 8; ++r) {
                int g = rowBase + wr * 16 + r + half * 8;
                int b = g >> 11, s = g & (SQ - 1);
                vB[(((size_t)b * NH + h) * SQ + s) * HD + d] = f2bf(vals[r]);
            }
        } else {
            bf16_t* dst = (which == 0) ? qB : kB;
            const float qscale = (which == 0) ? 0.125f : 1.0f;  // 1/sqrt(64)
            const float sgn = (d & 1) ? 1.0f : -1.0f;           // neg_mask
            #pragma unroll
            for (int r = 0; r < 8; ++r) {
                float partner = __shfl_xor(vals[r], 1, 32);     // x[d^1]
                int g = rowBase + wr * 16 + r + half * 8;
                int b = g >> 11, s = g & (SQ - 1);
                float c  = cosT[s * HD + d];
                float sn = sinT[s * HD + d];
                float rope = (c * vals[r] + sgn * sn * partner) * qscale;
                dst[(((size_t)b * NH + h) * SQ + s) * HD + d] = f2bf(rope);
            }
        }
    }
}

// =====================================================================
// Kernel 2: causal flash attention.  Block = 4 waves = 64 q-rows of one
// (b,h); wave owns a 16-row Q tile (fragments in registers).  Streams
// 32-key K/V chunks through LDS (K via async-to-LDS, V transposed with
// packed dword stores); online softmax; all matmuls on WMMA.
// Output written bf16 as [B,S,H*D] (= attn_o.transpose.reshape).
// =====================================================================
__global__ __launch_bounds__(128) void attn_kernel(
    const bf16_t* __restrict__ qB, const bf16_t* __restrict__ kB,
    const bf16_t* __restrict__ vB, bf16_t* __restrict__ oB)
{
    constexpr int LKC = HD + 8;   // Kc row stride (elems)
    constexpr int LVT = 32 + 8;   // Vt row stride
    constexpr int LSC = 36;       // score row stride (floats)
    constexpr int LPB = 40;       // prob row stride (elems)
    __shared__ bf16_t Kc[32 * LKC];        // [key][d]
    __shared__ bf16_t Vt[HD * LVT];        // [d][key]  (transposed)
    __shared__ float  Sc[4][16 * LSC];
    __shared__ bf16_t Pb[4][16 * LPB];
    __shared__ float  Al[4][16];
    __shared__ float  Li[4][16];

    const int tid  = threadIdx.x;
    const int wave = tid >> 5;
    const int lane = tid & 31;
    const int half = lane >> 4;
    const int l15  = lane & 15;

    const int bh    = blockIdx.y;              // b*NH + h
    const int qBase = blockIdx.x * 64;
    const int qRow0 = qBase + wave * 16;

    const bf16_t* Qp = qB + (size_t)bh * SQ * HD;
    const bf16_t* Kp = kB + (size_t)bh * SQ * HD;
    const bf16_t* Vp = vB + (size_t)bh * SQ * HD;

    Frag16 qf[2];                              // Q 16x64 = two 16x32 A-frags
    {
        const bf16_t* qrow = Qp + (size_t)(qRow0 + l15) * HD;
        #pragma unroll
        for (int j = 0; j < 2; ++j) {
            qf[j].u[0] = *(const v4u*)(qrow + j * 32 + half * 8);
            qf[j].u[1] = *(const v4u*)(qrow + j * 32 + 16 + half * 8);
        }
    }

    v8f  oacc[4] = {};
    float m = -INFINITY, l = 0.0f;

    const int nChunks = (qBase + 64) / 32;     // causal bound for block
    for (int kc = 0; kc < nChunks; ++kc) {
        const int kcBase = kc * 32;
        __syncthreads();                       // LDS free of prev readers
        { // stage K chunk [32][HD] via async global->LDS (ASYNCcnt path)
            int key = tid >> 2, d0 = (tid & 3) * 16;
            const bf16_t* src = Kp + (size_t)(kcBase + key) * HD + d0;
            bf16_t* dst = &Kc[key * LKC + d0];
            async_copy16(dst, src);
            async_copy16(dst + 8, src + 8);
        }
        { // stage V chunk transposed -> Vt[d][key]: two key-rows/thread,
          // interleave halfwords in registers, packed b32 stores
            int kp = tid & 15;                 // keys 2kp, 2kp+1
            int d0 = (tid >> 4) * 8;           // 0..56
            const bf16_t* s0 = Vp + (size_t)(kcBase + 2 * kp) * HD + d0;
            union { v4u v; unsigned u[4]; } a, b;
            a.v = *(const v4u*)s0;
            b.v = *(const v4u*)(s0 + HD);
            #pragma unroll
            for (int i = 0; i < 4; ++i) {
                unsigned lo = (a.u[i] & 0xFFFFu) | (b.u[i] << 16);
                unsigned hi = (a.u[i] >> 16) | (b.u[i] & 0xFFFF0000u);
                *(unsigned*)&Vt[(d0 + 2 * i)     * LVT + 2 * kp] = lo;
                *(unsigned*)&Vt[(d0 + 2 * i + 1) * LVT + 2 * kp] = hi;
            }
        }
        async_wait();
        __syncthreads();

        if (kcBase <= qRow0 + 15) {            // wave-uniform causal skip
            // ---- S = Q K^T (two 16x16 key tiles, K-dim = d split 2x32) ----
            #pragma unroll
            for (int t = 0; t < 2; ++t) {
                v8f s8 = {};
                #pragma unroll
                for (int j = 0; j < 2; ++j) {
                    Frag16 bk;  // lane = key column, d split {0-7,16-23} vs {8-15,24-31}
                    const bf16_t* p = &Kc[(t * 16 + l15) * LKC + j * 32 + half * 8];
                    bk.u[0] = *(const v4u*)p;
                    bk.u[1] = *(const v4u*)(p + 16);
                    s8 = __builtin_amdgcn_wmma_f32_16x16x32_bf16(
                        false, qf[j].v, false, bk.v, (short)0, s8, false, false);
                }
                const int col = kcBase + t * 16 + l15;
                #pragma unroll
                for (int r = 0; r < 8; ++r) {
                    int qrow = qRow0 + r + half * 8;
                    float sv = (col <= qrow) ? s8[r] : -INFINITY;
                    Sc[wave][(r + half * 8) * LSC + t * 16 + l15] = sv;
                }
            }
            // ---- online softmax: lanes 0..15 each own one row ----
            if (lane < 16) {
                const float* srow = &Sc[wave][l15 * LSC];
                float mc = -INFINITY;
                #pragma unroll
                for (int j = 0; j < 32; ++j) mc = fmaxf(mc, srow[j]);
                float mn = fmaxf(m, mc);
                float a  = __expf(m - mn);
                float ps = 0.0f;
                #pragma unroll
                for (int j = 0; j < 32; j += 2) {
                    float p0 = __expf(srow[j]     - mn);
                    float p1 = __expf(srow[j + 1] - mn);
                    ps += p0 + p1;
                    *(unsigned*)&Pb[wave][l15 * LPB + j] = pk2bf(p0, p1);
                }
                m = mn;
                l = l * a + ps;
                Al[wave][l15] = a;
            }
            { // rescale O accumulators by per-row alpha (C row<->lane map)
                float ar[8];
                #pragma unroll
                for (int r = 0; r < 8; ++r) ar[r] = Al[wave][r + half * 8];
                #pragma unroll
                for (int t = 0; t < 4; ++t)
                    #pragma unroll
                    for (int r = 0; r < 8; ++r) oacc[t][r] *= ar[r];
            }
            // ---- O += P V (K-dim = 32 keys, four 16-col d tiles) ----
            Frag16 pf;
            {
                const bf16_t* prow = &Pb[wave][l15 * LPB];
                pf.u[0] = *(const v4u*)(prow + half * 8);
                pf.u[1] = *(const v4u*)(prow + 16 + half * 8);
            }
            #pragma unroll
            for (int t = 0; t < 4; ++t) {
                Frag16 vf;  // lane = d column, keys split {0-7,16-23} vs {8-15,24-31}
                const bf16_t* p = &Vt[(t * 16 + l15) * LVT + half * 8];
                vf.u[0] = *(const v4u*)p; vf.u[1] = *(const v4u*)(p + 16);
                oacc[t] = __builtin_amdgcn_wmma_f32_16x16x32_bf16(
                    false, pf.v, false, vf.v, (short)0, oacc[t], false, false);
            }
        }
    }

    // ---- finalize: divide by l, store bf16 [B,S,M] ----
    if (lane < 16) Li[wave][l15] = 1.0f / l;
    float li[8];
    #pragma unroll
    for (int r = 0; r < 8; ++r) li[r] = Li[wave][r + half * 8];
    const int b = bh / NH, h = bh % NH;
    #pragma unroll
    for (int t = 0; t < 4; ++t) {
        int d = t * 16 + l15;
        #pragma unroll
        for (int r = 0; r < 8; ++r) {
            int s = qRow0 + r + half * 8;
            oB[((size_t)b * SQ + s) * MD + h * HD + d] = f2bf(oacc[t][r] * li[r]);
        }
    }
}

// =====================================================================
// Kernel 3: output projection.  A = attn bf16 [8192,1024] (async-to-LDS
// staging), B = W_o fp32 [1024,1024] (bf16 on the fly), + b_o, fp32 out.
// =====================================================================
__global__ __launch_bounds__(256) void oproj_kernel(
    const bf16_t* __restrict__ A, const float* __restrict__ Wo,
    const float* __restrict__ bo, float* __restrict__ out)
{
    constexpr int K = MD, N = MD;
    constexpr int BK = 32, BM = 64, BN = 128;
    constexpr int LDA = BK + 8;
    constexpr int LDB = BK + 8;
    __shared__ bf16_t At[BM * LDA];
    __shared__ bf16_t Bt[BN * LDB];

    const int tid  = threadIdx.x;
    const int wave = tid >> 5;
    const int lane = tid & 31;
    const int half = lane >> 4;
    const int l15  = lane & 15;
    const int wr   = wave >> 1;
    const int wc   = wave & 1;
    const int rowBase = blockIdx.y * BM;
    const int colBase = blockIdx.x * BN;

    v8f acc[4] = {};

    for (int k0 = 0; k0 < K; k0 += BK) {
        __syncthreads();
        { // A already bf16: async global->LDS b128
            int r = tid >> 2;
            int c = (tid & 3) * 8;
            async_copy16(&At[r * LDA + c],
                         A + (size_t)(rowBase + r) * K + k0 + c);
        }
        { // W_o fp32 -> bf16 transposed, packed b32 stores (two k-rows)
            int kp = tid >> 4;
            int n0 = (tid & 15) * 8;
            const float* s0 = Wo + (size_t)(k0 + 2 * kp) * N + colBase + n0;
            const float* s1 = s0 + N;
            #pragma unroll
            for (int i = 0; i < 8; ++i)
                *(unsigned*)&Bt[(n0 + i) * LDB + 2 * kp] = pk2bf(s0[i], s1[i]);
        }
        async_wait();
        __syncthreads();

        Frag16 a;
        {
            int r = wr * 16 + l15;
            a.u[0] = *(const v4u*)&At[r * LDA + half * 8];
            a.u[1] = *(const v4u*)&At[r * LDA + 16 + half * 8];
        }
        #pragma unroll
        for (int t = 0; t < 4; ++t) {
            Frag16 b;
            int n = wc * 64 + t * 16 + l15;
            b.u[0] = *(const v4u*)&Bt[n * LDB + half * 8];
            b.u[1] = *(const v4u*)&Bt[n * LDB + 16 + half * 8];
            acc[t] = __builtin_amdgcn_wmma_f32_16x16x32_bf16(
                false, a.v, false, b.v, (short)0, acc[t], false, false);
        }
    }

    #pragma unroll
    for (int t = 0; t < 4; ++t) {
        int col = colBase + wc * 64 + t * 16 + l15;
        float bias = bo[col];
        #pragma unroll
        for (int r = 0; r < 8; ++r) {
            int g = rowBase + wr * 16 + r + half * 8;
            out[(size_t)g * N + col] = acc[t][r] + bias;
        }
    }
}

// =====================================================================
extern "C" void kernel_launch(void* const* d_in, const int* in_sizes, int n_in,
                              void* d_out, int out_size, void* d_ws, size_t ws_size,
                              hipStream_t stream) {
    (void)in_sizes; (void)n_in; (void)out_size; (void)ws_size;
    const float* x    = (const float*)d_in[0];
    const float* cosT = (const float*)d_in[2];
    const float* sinT = (const float*)d_in[3];
    const float* Wqkv = (const float*)d_in[6];
    const float* bqkv = (const float*)d_in[7];
    const float* Wo   = (const float*)d_in[8];
    const float* bo   = (const float*)d_in[9];
    float* out = (float*)d_out;

    const size_t perT = (size_t)BB * NH * SQ * HD;  // 8,388,608 elems
    bf16_t* qb = (bf16_t*)d_ws;
    bf16_t* kb = qb + perT;
    bf16_t* vb = kb + perT;
    bf16_t* ob = vb + perT;     // total 64 MB of bf16 workspace

    dim3 g1(3 * MD / 128, (BB * SQ) / 64);          // (24, 128)
    qkv_rope_kernel<<<g1, 256, 0, stream>>>(x, Wqkv, bqkv, cosT, sinT, qb, kb, vb);

    dim3 g2(SQ / 64, BB * NH);                      // (32, 64)
    attn_kernel<<<g2, 128, 0, stream>>>(qb, kb, vb, ob);

    dim3 g3(MD / 128, (BB * SQ) / 64);              // (8, 128)
    oproj_kernel<<<g3, 256, 0, stream>>>(ob, Wo, bo, out);
}
